// RotationEquivariantConv2d_65738769432738
// MI455X (gfx1250) — matrix operand order repaired and checked by reading, not compiled, not run on hardware
//
#include <hip/hip_runtime.h>

// ---------------------------------------------------------------------------
// CDNA5 (gfx1250) rotation-equivariant conv.
//  - implicit GEMM on v_wmma_f32_16x16x32_bf16 (wave32)
//  - rotations folded into GEMM M (m = o*8 + r) -> max over rotations is a
//    per-lane max over the 8 accumulator VGPRs of each 16x16 D tile
//  - A-tiles (pre-rotated bf16 weights, fragment-ready [t][m][k] layout) are
//    DMA'd into LDS with the Tensor Data Mover (tensor_load_to_lds,
//    TENSORcnt), one 3-D tile (k=32, m=128, t=9) per 32-channel chunk.
// ---------------------------------------------------------------------------

typedef __attribute__((ext_vector_type(16))) __bf16        v16bf;
typedef __attribute__((ext_vector_type(8)))  float         v8f;
typedef __attribute__((ext_vector_type(4)))  unsigned int  v4u;
typedef __attribute__((ext_vector_type(8)))  int           v8i;
typedef __attribute__((ext_vector_type(4)))  int           v4i;

__device__ __forceinline__ unsigned short f2bf(float f) {
    unsigned int u = __float_as_uint(f);
    unsigned int r = u + 0x7FFFu + ((u >> 16) & 1u);   // round-to-nearest-even
    return (unsigned short)(r >> 16);
}

// ---------------------------------------------------------------------------
// Pre-pass: bilinear-rotate the 3x3 filters (align_corners=False, matching
// torch affine_grid/grid_sample) into bf16 weights, fragment-ready layout:
//   rwT[((t*2048 + m)*256 + i)],  t = ky*3+kx (tap),  m = o*8 + r.
// ---------------------------------------------------------------------------
__global__ __launch_bounds__(256) void rotw_kernel(
    const float* __restrict__ w, unsigned short* __restrict__ rwT)
{
    const int gid = blockIdx.x * 256 + threadIdx.x;     // 9*2048*256 total
    const int i = gid & 255;
    const int m = (gid >> 8) & 2047;
    const int t = gid >> 19;                            // 0..8
    const int r = m & 7;
    const int o = m >> 3;
    const int ky = t / 3, kx = t % 3;

    // pixel-center grid coords in [-1,1] (k=3)
    const float cx = (2.0f * (float)kx + 1.0f) * (1.0f / 3.0f) - 1.0f;
    const float cy = (2.0f * (float)ky + 1.0f) * (1.0f / 3.0f) - 1.0f;
    const float ang = 0.7853981633974483f * (float)r;   // 2*pi*r/8
    const float cs = __cosf(ang), sn = __sinf(ang);
    const float xs = cs * cx - sn * cy;
    const float ys = sn * cx + cs * cy;
    // unnormalize (align_corners=False): ix = ((x+1)*k - 1)/2
    const float ix = ((xs + 1.0f) * 3.0f - 1.0f) * 0.5f;
    const float iy = ((ys + 1.0f) * 3.0f - 1.0f) * 0.5f;
    const float ix0f = floorf(ix), iy0f = floorf(iy);
    const float wx1 = ix - ix0f, wy1 = iy - iy0f;
    const float wx0 = 1.0f - wx1, wy0 = 1.0f - wy1;
    const int ix0 = (int)ix0f, iy0 = (int)iy0f;
    const int ix1 = ix0 + 1,   iy1 = iy0 + 1;

    const float* wb = w + (size_t)(o * 256 + i) * 9;
    auto g = [&](int yy, int xx) -> float {
        return ((unsigned)yy < 3u && (unsigned)xx < 3u) ? wb[yy * 3 + xx] : 0.0f;
    };
    const float acc = g(iy0, ix0) * (wy0 * wx0) + g(iy0, ix1) * (wy0 * wx1)
                    + g(iy1, ix0) * (wy1 * wx0) + g(iy1, ix1) * (wy1 * wx1);

    rwT[gid] = f2bf(acc);
}

// ---------------------------------------------------------------------------
// Main conv kernel.  Block = 256 thr (8 wave32).  Tile: 128 M-rows
// (16 out-ch x 8 rot) x 64 pixels.  K = 2304 walked as 8 chunks of 32
// channels; x patch staged once per chunk (f32->bf16, channel-fastest);
// all 9 taps of the A tile TDM'd into LDS per chunk.
// ---------------------------------------------------------------------------
#define LDS_B_BYTES 13056          // 3*68*32 bf16
#define LDS_A_BYTES 73728          // 9*128*32 bf16
#define LDS_TOTAL   (LDS_B_BYTES + LDS_A_BYTES)

__global__ __launch_bounds__(256) void conv_rot_wmma_kernel(
    const float* __restrict__ x,
    const unsigned short* __restrict__ rwT,
    float* __restrict__ out)
{
    // 64B-aligned dynamic LDS so fragment reads can use ds_load_b128
    extern __shared__ __align__(64) unsigned char smem[];
    unsigned int* lB =
        (unsigned int*)__builtin_assume_aligned(smem, 64);                 // [3][68][32] bf16
    unsigned short* sB16 =
        (unsigned short*)__builtin_assume_aligned(smem, 64);
    unsigned int* lA =
        (unsigned int*)__builtin_assume_aligned(smem + LDS_B_BYTES, 64);   // [9][128][32] bf16
    unsigned short* sA16 =
        (unsigned short*)__builtin_assume_aligned(smem + LDS_B_BYTES, 64);

    const int tid     = threadIdx.x;
    const int wave    = tid >> 5;
    const int lane    = tid & 31;
    const int n       = lane & 15;          // column within 16-wide tile
    const int halfsel = lane >> 4;          // 0: lanes 0-15, 1: lanes 16-31

    const int pt  = blockIdx.x;             // 0..1023 pixel tiles
    const int ot  = blockIdx.y;             // 0..15 M tiles (128 rows each)
    const int b   = pt >> 8;                // 256 tiles per batch image
    const int rem = pt & 255;
    const int h   = rem >> 1;
    const int w0  = (rem & 1) << 6;
    const int m0  = ot << 7;

    const int mrow = (wave << 4) + n;       // A-fragment row (0..127)

    v8f acc[4];
#pragma unroll
    for (int nt = 0; nt < 4; ++nt) acc[nt] = (v8f)0.0f;

    for (int ic0 = 0; ic0 < 256; ic0 += 32) {
        __syncthreads();   // previous chunk's LDS reads complete

        // ---- stage x patch: ch c in [ic0,ic0+32), rows h-1..h+1, px w0-1..w0+64
        for (int idx = tid; idx < 32 * 3 * 68; idx += 256) {
            const int pp = idx % 68;
            const int rr = (idx / 68) % 3;
            const int c  = idx / 204;
            const int hh = h + rr - 1;
            const int ww = w0 + pp - 1;
            float v = 0.0f;
            if (pp < 66 && (unsigned)hh < 128u && (unsigned)ww < 128u)
                v = x[((((b << 8) + ic0 + c) << 7) + hh) * 128 + ww];
            sB16[(rr * 68 + pp) * 32 + c] = f2bf(v);
            if (ic0 + 32 < 256)  // hint next chunk into cache (global_prefetch_b8)
                __builtin_prefetch(&x[((((b << 8) + ic0 + 32 + c) << 7) + h) * 128 + w0], 0, 3);
        }

        // ---- stage ALL 9 A tiles for this chunk: LDS <- rwT 3-D tile via TDM
        //      tile (k=32, m=128, t=9); strides 1 / 256 / 2048*256 (bf16 units)
#if __has_builtin(__builtin_amdgcn_tensor_load_to_lds)
        if (wave == 0) {
            const unsigned long long ga =
                (unsigned long long)(const void*)(rwT + (size_t)m0 * 256 + ic0);
            const unsigned int la = (unsigned int)(unsigned long long)(const void*)sA16;
            v4u g0; v8i g1; v4i g2; v4i g3;
            // group0: count=1 | lds_addr | global_addr[56:0] | type=2
            g0[0] = 1u;
            g0[1] = la;
            g0[2] = (unsigned int)(ga & 0xFFFFFFFFull);
            g0[3] = (unsigned int)((ga >> 32) & 0x01FFFFFFull) | (2u << 30);
            // group1: data_size=1(2B); tensor_dim0=256; tensor_dim1=2048;
            //         tile_dim0=32; tile_dim1=128; tile_dim2=9;
            //         dim0_stride=256; dim1_stride=2048*256
            g1[0] = (int)(1u << 16);
            g1[1] = (int)(256u << 16);
            g1[2] = (int)(2048u << 16);
            g1[3] = (int)(32u << 16);
            g1[4] = (int)(128u | (9u << 16));
            g1[5] = (int)256;
            g1[6] = 0;
            g1[7] = (int)((2048u * 256u) >> 16);
            // group2: tensor_dim2=9; dim2_stride=2048*256; tile_dim3=0
            g2[0] = 9; g2[1] = 0; g2[2] = (int)(2048u * 256u); g2[3] = 0;
            g3[0] = 0; g3[1] = 0; g3[2] = 0; g3[3] = 0;
#if __clang_major__ >= 23
            v8i g4 = (v8i)0;
            __builtin_amdgcn_tensor_load_to_lds(g0, g1, g2, g3, g4, 0);
#else
            __builtin_amdgcn_tensor_load_to_lds(g0, g1, g2, g3, 0);
#endif
            __builtin_amdgcn_s_wait_tensorcnt(0);
        }
#else
        for (int idx = tid; idx < 9 * 4096; idx += 256) {
            const int k = idx & 31;
            const int m = (idx >> 5) & 127;
            const int t = idx >> 12;
            sA16[(t * 128 + m) * 32 + k] =
                rwT[((size_t)(t * 2048 + m0 + m)) * 256 + ic0 + k];
        }
#endif
        __syncthreads();   // A (TDM) + B staged, visible to all waves

#pragma unroll
        for (int t = 0; t < 9; ++t) {
            // ---- A fragment (16x32 bf16): half selects K 0-7/16-23 vs 8-15/24-31
            const int abase = (t * 128 + mrow) * 16 + (halfsel << 2);
            union { unsigned int u[8]; v16bf v; } af;
#pragma unroll
            for (int j = 0; j < 4; ++j) {
                af.u[j]     = lA[abase + j];
                af.u[4 + j] = lA[abase + 8 + j];
            }

            const int dy = t / 3;          // patch row
            const int dx = t % 3;          // px shift (-1..+1 folded into pad)
#pragma unroll
            for (int nt = 0; nt < 4; ++nt) {
                const int p = nt * 16 + n + dx;                 // 0..65
                const int bbase = (dy * 68 + p) * 16 + (halfsel << 2);
                union { unsigned int u[8]; v16bf v; } bf;
#pragma unroll
                for (int j = 0; j < 4; ++j) { bf.u[j] = lB[bbase + j]; bf.u[4 + j] = lB[bbase + 8 + j]; }

                acc[nt] = __builtin_amdgcn_wmma_f32_16x16x32_bf16(
                    /*neg_a=*/false, af.v, /*neg_b=*/false, bf.v,
                    /*c_mod=*/(short)0, acc[nt],
                    /*reuse_a=*/false, /*reuse_b=*/false);
            }
        }
    }

    // ---- max over the 8 rotations (8 accumulator VGPRs = 8 M-rows = 8 r) ----
    const int o = (ot << 4) + (wave << 1) + halfsel;
#pragma unroll
    for (int nt = 0; nt < 4; ++nt) {
        float mx = acc[nt][0];
#pragma unroll
        for (int v = 1; v < 8; ++v) mx = fmaxf(mx, acc[nt][v]);
        out[((((b << 8) + o) << 7) + h) * 128 + w0 + nt * 16 + n] = mx;
    }
}

// ---------------------------------------------------------------------------
extern "C" void kernel_launch(void* const* d_in, const int* in_sizes, int n_in,
                              void* d_out, int out_size, void* d_ws, size_t ws_size,
                              hipStream_t stream) {
    (void)in_sizes; (void)n_in; (void)out_size; (void)ws_size;
    const float* x = (const float*)d_in[0];        // (4,256,128,128) f32
    const float* w = (const float*)d_in[1];        // (256,256,3,3)  f32
    float* out = (float*)d_out;                    // (4,256,128,128) f32
    unsigned short* rwT = (unsigned short*)d_ws;   // 9*2048*256 bf16 = 9.44 MB

    // 1) bake rotated bf16 weights (fragment-ready [t][m][k] layout)
    rotw_kernel<<<(9 * 2048 * 256) / 256, 256, 0, stream>>>(w, rwT);

    // 2) implicit-GEMM conv + max over rotations (TDM-staged A tiles)
    dim3 grid(1024, 16);   // 1024 pixel tiles x 16 M tiles
    conv_rot_wmma_kernel<<<grid, 256, LDS_TOTAL, stream>>>(x, rwT, out);
}
